// RoutableLmModel_51256139710798
// MI455X (gfx1250) — matrix operand order repaired and checked by reading, not compiled, not run on hardware
//
#include <hip/hip_runtime.h>
#include <hip/hip_bf16.h>

// MoE router: gather -> bf16 WMMA GEMM (fp32 accum) -> top4 + softmax -> expert mask
// Shapes fixed by the reference: B=8, S=4096, D=2048, E=64, TOP_K=4.
//
// v2: each wave computes a 32-row M-tile (two 16-row WMMA tiles) so every
// B-fragment load feeds two v_wmma_f32_16x16x32_bf16 ops: 16 b128 loads per
// 8 WMMAs (was 12 per 4), halving L2-side router-weight traffic to ~256 MB
// and matching the ~256 MB HBM floor of the gathered activations.

typedef __attribute__((ext_vector_type(16))) __bf16 v16bf;
typedef __attribute__((ext_vector_type(8)))  float  v8f;
typedef __attribute__((ext_vector_type(8)))  int    v8i;

#define B_  8
#define S_  4096
#define D_  2048
#define E_  64
#define KT_ (D_ / 32)   // 64 k-tiles of 32
#define NT_ (E_ / 16)   // 4 n-tiles of 16

// ---------------------------------------------------------------------------
// Pre-pass: repack router_w (fp32, [D,E] row-major over K) into bf16 laid out
// in WMMA B-fragment order.  Fragment (kt, j) = B tile K=kt*32..+31, N=16j..+15.
// Per-lane layout (16-bit B 32x16): lane n (0-15) element e holds
// B[kt*32 + e + 8*(e/8)][16j + n]; lane n+16 element e holds B[kt*32 + 8 + e + 8*(e/8)][16j + n].
// Storage: wp[((kt*NT_ + j)*32 + lane)*8 + d] packs elements 2d, 2d+1.
// ---------------------------------------------------------------------------
__global__ void pack_router_w_kernel(const float* __restrict__ w,
                                     unsigned int* __restrict__ wp) {
  int tid  = blockIdx.x * blockDim.x + threadIdx.x;   // one packed dword each
  int d    = tid & 7;
  int lane = (tid >> 3) & 31;
  int j    = (tid >> 8) & 3;
  int kt   = tid >> 10;
  if (kt >= KT_) return;
  int n  = 16 * j + (lane & 15);
  int k0 = kt * 32 + ((lane >> 4) << 3) + 2 * d + ((d >> 2) << 3);
  __bf16 b0 = (__bf16)w[(size_t)k0 * E_ + n];
  __bf16 b1 = (__bf16)w[(size_t)(k0 + 1) * E_ + n];
  unsigned int u = ((unsigned int)__builtin_bit_cast(unsigned short, b1) << 16) |
                    (unsigned int)__builtin_bit_cast(unsigned short, b0);
  wp[tid] = u;
}

// Build one A fragment (16-bit A 16x32 layout) from a gathered fp32 row.
// lane<16 needs K = k0+0..7 and k0+16..23; lane>=16 is shifted by +8.
__device__ __forceinline__ v16bf load_a_frag(const float* __restrict__ arow,
                                             int k0) {
  float4 f0 = *(const float4*)(arow + k0);
  float4 f1 = *(const float4*)(arow + k0 + 4);
  float4 f2 = *(const float4*)(arow + k0 + 16);
  float4 f3 = *(const float4*)(arow + k0 + 20);
  v16bf a;
  a[0]  = (__bf16)f0.x; a[1]  = (__bf16)f0.y; a[2]  = (__bf16)f0.z; a[3]  = (__bf16)f0.w;
  a[4]  = (__bf16)f1.x; a[5]  = (__bf16)f1.y; a[6]  = (__bf16)f1.z; a[7]  = (__bf16)f1.w;
  a[8]  = (__bf16)f2.x; a[9]  = (__bf16)f2.y; a[10] = (__bf16)f2.z; a[11] = (__bf16)f2.w;
  a[12] = (__bf16)f3.x; a[13] = (__bf16)f3.y; a[14] = (__bf16)f3.z; a[15] = (__bf16)f3.w;
  return a;
}

// ---------------------------------------------------------------------------
// Main kernel: 8 waves/block, one 32-row M-tile per wave (two 16-row subtiles
// sharing the B fragments).
// ---------------------------------------------------------------------------
__global__ void __launch_bounds__(256)
moe_router_wmma_kernel(const float* __restrict__ hs,
                       const int*   __restrict__ idxs,
                       const unsigned int* __restrict__ wp,
                       float* __restrict__ out) {
  __shared__ float logits_lds[8][32][E_];   // 64 KB: per-wave 32x64 logits tile

  const int lane = threadIdx.x & 31;
  const int wave = threadIdx.x >> 5;
  const int hi   = lane >> 4;        // 0: lanes 0-15, 1: lanes 16-31
  const int m    = lane & 15;

  const int tile    = blockIdx.x * 8 + wave;     // 1024 tiles of 32 rows
  const int rowBase = tile * 32;                 // flat (b*S + s) row base
  const int fr0     = rowBase + m;               // subtile 0: rows 0..15
  const int fr1     = rowBase + 16 + m;          // subtile 1: rows 16..31
  const int src0    = idxs[fr0];
  const int src1    = idxs[fr1];
  const float* arow0 = hs + ((size_t)((fr0 >> 12) << 12) + (size_t)src0) * D_;
  const float* arow1 = hs + ((size_t)((fr1 >> 12) << 12) + (size_t)src1) * D_;

  v8f accA0 = {}, accA1 = {}, accA2 = {}, accA3 = {};   // rows 0..15
  v8f accB0 = {}, accB1 = {}, accB2 = {}, accB3 = {};   // rows 16..31
  const v8i* wpt = (const v8i*)wp;                      // 32B B-fragment per lane

#pragma unroll 2
  for (int kt = 0; kt < KT_; ++kt) {
    const int k0 = kt * 32 + hi * 8;
    v16bf a0 = load_a_frag(arow0, k0);
    v16bf a1 = load_a_frag(arow1, k0);

    const v8i* bp = wpt + (size_t)(kt * NT_) * 32 + lane;
    v16bf bf0 = __builtin_bit_cast(v16bf, bp[0]);
    v16bf bf1 = __builtin_bit_cast(v16bf, bp[32]);
    v16bf bf2 = __builtin_bit_cast(v16bf, bp[64]);
    v16bf bf3 = __builtin_bit_cast(v16bf, bp[96]);

    accA0 = __builtin_amdgcn_wmma_f32_16x16x32_bf16(false, a0, false, bf0, (short)0, accA0, false, false);
    accB0 = __builtin_amdgcn_wmma_f32_16x16x32_bf16(false, a1, false, bf0, (short)0, accB0, false, false);
    accA1 = __builtin_amdgcn_wmma_f32_16x16x32_bf16(false, a0, false, bf1, (short)0, accA1, false, false);
    accB1 = __builtin_amdgcn_wmma_f32_16x16x32_bf16(false, a1, false, bf1, (short)0, accB1, false, false);
    accA2 = __builtin_amdgcn_wmma_f32_16x16x32_bf16(false, a0, false, bf2, (short)0, accA2, false, false);
    accB2 = __builtin_amdgcn_wmma_f32_16x16x32_bf16(false, a1, false, bf2, (short)0, accB2, false, false);
    accA3 = __builtin_amdgcn_wmma_f32_16x16x32_bf16(false, a0, false, bf3, (short)0, accA3, false, false);
    accB3 = __builtin_amdgcn_wmma_f32_16x16x32_bf16(false, a1, false, bf3, (short)0, accB3, false, false);
  }

  // C layout (32-bit 16x16): VGPR r, lanes 0-15 -> (M=r, N=lane), lanes 16-31
  // -> (M=r+8, N=lane-16).  Spill to LDS to transpose rows for the epilogue.
#pragma unroll
  for (int r = 0; r < 8; ++r) {
    const int row = r + hi * 8;
    logits_lds[wave][row][ 0 + m] = accA0[r];
    logits_lds[wave][row][16 + m] = accA1[r];
    logits_lds[wave][row][32 + m] = accA2[r];
    logits_lds[wave][row][48 + m] = accA3[r];
    logits_lds[wave][row + 16][ 0 + m] = accB0[r];
    logits_lds[wave][row + 16][16 + m] = accB1[r];
    logits_lds[wave][row + 16][32 + m] = accB2[r];
    logits_lds[wave][row + 16][48 + m] = accB3[r];
  }
  __syncthreads();

  // All 32 lanes: one output row each.  Top-4 (strict '>' keeps the earliest
  // index on ties, matching jax.lax.top_k), softmax over the 4 values.
  {
    const float* lr = logits_lds[wave][lane];
    float t0 = -3.4e38f, t1 = -3.4e38f, t2 = -3.4e38f, t3 = -3.4e38f;
    int   i0 = 0, i1 = 0, i2 = 0, i3 = 0;
#pragma unroll 8
    for (int e = 0; e < E_; ++e) {
      float v = lr[e];
      if (v > t0)      { t3 = t2; i3 = i2; t2 = t1; i2 = i1; t1 = t0; i1 = i0; t0 = v; i0 = e; }
      else if (v > t1) { t3 = t2; i3 = i2; t2 = t1; i2 = i1; t1 = v;  i1 = e; }
      else if (v > t2) { t3 = t2; i3 = i2; t2 = v;  i2 = e; }
      else if (v > t3) { t3 = v;  i3 = e; }
    }
    float w0 = 1.0f;
    float w1 = __expf(t1 - t0);
    float w2 = __expf(t2 - t0);
    float w3 = __expf(t3 - t0);
    float inv = 1.0f / (w0 + w1 + w2 + w3);
    w0 *= inv; w1 *= inv; w2 *= inv; w3 *= inv;

    float4* orow = (float4*)(out + (size_t)(rowBase + lane) * E_);
#pragma unroll
    for (int c = 0; c < 16; ++c) {
      const int e0 = c * 4;
      float4 o;
      o.x = (i0 == e0    ) ? w0 : (i1 == e0    ) ? w1 : (i2 == e0    ) ? w2 : (i3 == e0    ) ? w3 : 0.0f;
      o.y = (i0 == e0 + 1) ? w0 : (i1 == e0 + 1) ? w1 : (i2 == e0 + 1) ? w2 : (i3 == e0 + 1) ? w3 : 0.0f;
      o.z = (i0 == e0 + 2) ? w0 : (i1 == e0 + 2) ? w1 : (i2 == e0 + 2) ? w2 : (i3 == e0 + 2) ? w3 : 0.0f;
      o.w = (i0 == e0 + 3) ? w0 : (i1 == e0 + 3) ? w1 : (i2 == e0 + 3) ? w2 : (i3 == e0 + 3) ? w3 : 0.0f;
      orow[c] = o;
    }
  }
}

extern "C" void kernel_launch(void* const* d_in, const int* in_sizes, int n_in,
                              void* d_out, int out_size, void* d_ws, size_t ws_size,
                              hipStream_t stream) {
  const float* hs  = (const float*)d_in[0];        // [B,S,D] fp32
  const int*   idx = (const int*)  d_in[1];        // [B,S] int32
  const float* rw  = (const float*)d_in[2];        // [D,E] fp32
  float*       out = (float*)d_out;                // [B,S,E] fp32
  unsigned int* wp = (unsigned int*)d_ws;          // 64*4*32*8 dwords = 256 KB

  // Repack router weights into bf16 WMMA-B fragment order (65536 dwords).
  pack_router_w_kernel<<<256, 256, 0, stream>>>(rw, wp);

  // 1024 M-tiles of 32 rows; 8 waves (tiles) per 256-thread block.
  moe_router_wmma_kernel<<<128, 256, 0, stream>>>(hs, idx, wp, out);
}
